// NewtonNet_180388627172
// MI455X (gfx1250) — matrix operand order repaired and checked by reading, not compiled, not run on hardware
//
#include <hip/hip_runtime.h>

// ---------- types ----------
typedef __attribute__((ext_vector_type(16))) _Float16 v16h;
typedef __attribute__((ext_vector_type(8)))  _Float16 h8;
typedef __attribute__((ext_vector_type(8)))  float    v8f;

// fast silu: x * rcp(1+exp(-x))  (v_exp_f32 + v_add + v_rcp_f32 + v_mul, all TRANS/VALU co-exec friendly)
__device__ __forceinline__ float silu_f(float x) {
    return x * __builtin_amdgcn_rcpf(1.f + __expf(-x));
}

// ---------- WMMA GEMM: out[M,128] = epilogue(A[M,128] @ W[128,128] + bias) ----------
// ADT: 0 = A is f32, 1 = A is f16
// EPI: 0 = f32 out, 1 = f16 silu out, 2 = f32 out = addend + acc*mul, 3 = f16 plain out
// Wt is the transposed weight, f16, Wt[n*128 + k] = W[k][n].
template<int ADT, int EPI>
__global__ __launch_bounds__(256) void gemm128(
    const void* __restrict__ Av, const _Float16* __restrict__ Wt,
    const float* __restrict__ bias, void* __restrict__ outv,
    const float* __restrict__ addend, const float* __restrict__ mul)
{
    __shared__ _Float16 lds[16 * 136];           // 16 rows, padded stride (16B-aligned rows)
    const int t  = threadIdx.x;
    const int m0 = blockIdx.x * 16;

    // Stage A tile (16 x 128) into LDS as f16 (coalesced, 8 elems/thread)
    if (ADT == 0) {
        const float* A = (const float*)Av;
#pragma unroll
        for (int k = 0; k < 8; k++) {
            int idx = t + k * 256, r = idx >> 7, c = idx & 127;
            lds[r * 136 + c] = (_Float16)A[(m0 + r) * 128 + c];
        }
    } else {
        const _Float16* A = (const _Float16*)Av;
#pragma unroll
        for (int k = 0; k < 8; k++) {
            int idx = t + k * 256, r = idx >> 7, c = idx & 127;
            lds[r * 136 + c] = A[(m0 + r) * 128 + c];
        }
    }
    __syncthreads();

    const int lane = t & 31;
    const int n0   = (t >> 5) * 16;              // each wave: one 16-col slice
    const int ra   = lane & 15;                  // A row within tile
    const int ka   = (lane >> 4) * 16;           // A K-offset (contiguous k = 16h + s)
    const int cb   = n0 + (lane & 15);           // B column (= output col)
    const int kb   = (lane >> 4) * 16;           // B K-offset (contiguous k = 16h + s, same as A)

    float bv = bias ? bias[cb] : 0.f;
    v8f acc;
#pragma unroll
    for (int i = 0; i < 8; i++) acc[i] = bv;

    const _Float16* wrow = Wt + cb * 128;
    const _Float16* arow = lds + ra * 136;
    union U { v16h v; h8 h[2]; };
#pragma unroll
    for (int kc = 0; kc < 128; kc += 32) {
        U af, bf;
        af.h[0] = *(const h8*)(arow + kc + ka);
        af.h[1] = *(const h8*)(arow + kc + ka + 8);
        bf.h[0] = *(const h8*)(wrow + kc + kb);
        bf.h[1] = *(const h8*)(wrow + kc + kb + 8);
        acc = __builtin_amdgcn_wmma_f32_16x16x32_f16(
            false, af.v, false, bf.v, (short)0, acc, false, false);
    }

    const int moff = (lane >> 4) * 8;            // C layout: lanes 16-31 hold M+8
#pragma unroll
    for (int vv = 0; vv < 8; vv++) {
        int row = m0 + vv + moff;
        int o   = row * 128 + cb;
        float x = acc[vv];
        if (EPI == 0)       ((float*)outv)[o]     = x;
        else if (EPI == 1)  ((_Float16*)outv)[o]  = (_Float16)silu_f(x);
        else if (EPI == 2)  ((float*)outv)[o]     = addend[o] + x * mul[o];
        else                ((_Float16*)outv)[o]  = (_Float16)x;
    }
}

// ---------- weight convert + transpose (128x128 f32 -> f16 transposed) ----------
__global__ void cvtT(const float* __restrict__ W, _Float16* __restrict__ Wt) {
    int idx = blockIdx.x * 256 + threadIdx.x;    // 16384 elems
    int k = idx >> 7, n = idx & 127;
    Wt[n * 128 + k] = (_Float16)W[k * 128 + n];
}

__global__ void zero_f32(float* p, int n) {
    int i = blockIdx.x * 256 + threadIdx.x;
    if (i < n) p[i] = 0.f;
}

__global__ void embed_kernel(const int* __restrict__ Z, const float* __restrict__ emb,
                             float* __restrict__ a) {
    int i = blockIdx.x * 256 + threadIdx.x;      // NA*128
    int gi = i >> 7, f = i & 127;
    a[i] = emb[Z[gi] * 128 + f];
}

// ---------- fused RBF + msij + fW-dot + f_dir (one block per atom) ----------
__global__ __launch_bounds__(256) void msij_kernel(
    const float* __restrict__ D, const float* __restrict__ V,
    const int* __restrict__ N, const float* __restrict__ NM,
    const float* __restrict__ a_m, const float* __restrict__ rbfW,
    const float* __restrict__ rbfb, const float* __restrict__ fW,
    _Float16* __restrict__ msij, float* __restrict__ fscalar,
    float* __restrict__ fdir, int na)
{
    __shared__ float sines[64 * 20];
    __shared__ float cuts[64];
    __shared__ float red[256];
    __shared__ float fpart[6];
    const int t = threadIdx.x, gi = blockIdx.x;
    const int ebase = gi * 64;

    for (int idx = t; idx < 1280; idx += 256) {
        int n = idx / 20, r = idx % 20;
        float x = D[ebase + n];
        sines[idx] = __sinf((float)(r + 1) * 3.14159265358979f * x * 0.2f)
                   * __builtin_amdgcn_rcpf(x);
    }
    if (t < 64) {
        float x = D[ebase + t] * 0.2f;
        float x2 = x * x, x4 = x2 * x2, x9 = x4 * x4 * x;
        cuts[t] = (x < 1.f) ? (1.f - 55.f * x9 + 99.f * x9 * x - 45.f * x9 * x2) : 0.f;
    }
    __syncthreads();

    const int f = t & 127, g = t >> 7;
    const int batch = gi / na;
    float fa = 0.f;
    for (int pass = 0; pass < 32; pass++) {
        int n = pass * 2 + g;
        int e = ebase + n;
        float rm = rbfb[f];
#pragma unroll
        for (int r = 0; r < 20; r++) rm += sines[n * 20 + r] * rbfW[r * 128 + f];
        rm *= cuts[n];
        int j = batch * na + N[e];
        float ms = a_m[gi * 128 + f] * a_m[j * 128 + f] * rm;
        msij[e * 128 + f] = (_Float16)ms;
        red[t] = ms * fW[f];
        __syncthreads();
        for (int s = 64; s > 0; s >>= 1) {
            if ((t & 127) < s) red[t] += red[t + s];
            __syncthreads();
        }
        if ((t & 127) == 0) fscalar[e] = red[t];
        if (t < 6) {
            int gg = t / 3, d = t % 3;
            int ee = ebase + pass * 2 + gg;
            fa += red[gg * 128] * V[ee * 3 + d] * NM[ee];
        }
        __syncthreads();
    }
    if (t < 6) fpart[t] = fa;
    __syncthreads();
    if (t < 3) fdir[gi * 3 + t] += fpart[t] + fpart[t + 3];
}

// ---------- F_i = sum_n fs * (fscalar*V) * NM ; f_dyn += F_i ----------
__global__ __launch_bounds__(256) void fi_kernel(
    const _Float16* __restrict__ fs, const float* __restrict__ fscalar,
    const float* __restrict__ V, const float* __restrict__ NM,
    float* __restrict__ Fi, float* __restrict__ f_dyn)
{
    __shared__ float red[256];
    const int t = threadIdx.x, gi = blockIdx.x;
    const int f = t & 127, g = t >> 7;
    float a0 = 0, a1 = 0, a2 = 0;
    for (int n = g; n < 64; n += 2) {
        int e = gi * 64 + n;
        float c = (float)fs[e * 128 + f] * fscalar[e] * NM[e];
        a0 += c * V[e * 3 + 0]; a1 += c * V[e * 3 + 1]; a2 += c * V[e * 3 + 2];
    }
    float accs[3] = {a0, a1, a2};
    for (int d = 0; d < 3; d++) {
        red[t] = accs[d];
        __syncthreads();
        if (t < 128) {
            float tot = red[t] + red[t + 128];
            Fi[gi * 384 + d * 128 + f] = tot;
            f_dyn[gi * 384 + d * 128 + f] += tot;
        }
        __syncthreads();
    }
}

// ---------- dr_ext gather + r_dyn update + de ----------
__global__ __launch_bounds__(256) void drext_kernel(
    const _Float16* __restrict__ re, const int* __restrict__ N,
    const float* __restrict__ NM, const float* __restrict__ rold,
    const float* __restrict__ phir, const float* __restrict__ Fi,
    const float* __restrict__ f_dyn, float* __restrict__ rnew,
    float* __restrict__ de, int na)
{
    __shared__ float red[256];
    const int t = threadIdx.x, gi = blockIdx.x;
    const int f = t & 127, g = t >> 7;
    const int batch = gi / na;
    float a0 = 0, a1 = 0, a2 = 0;
    for (int n = g; n < 64; n += 2) {
        int e = gi * 64 + n;
        int j = batch * na + N[e];
        float w = (float)re[e * 128 + f] * NM[e];
        a0 += w * rold[j * 384 + 0 * 128 + f];
        a1 += w * rold[j * 384 + 1 * 128 + f];
        a2 += w * rold[j * 384 + 2 * 128 + f];
    }
    float accs[3] = {a0, a1, a2};
    float deacc = 0.f;
    for (int d = 0; d < 3; d++) {
        red[t] = accs[d];
        __syncthreads();
        if (t < 128) {
            float tot = red[t] + red[t + 128];
            float rn = rold[gi * 384 + d * 128 + f]
                     + phir[gi * 128 + f] * Fi[gi * 384 + d * 128 + f] + tot;
            rnew[gi * 384 + d * 128 + f] = rn;
            deacc -= f_dyn[gi * 384 + d * 128 + f] * rn;
        }
        __syncthreads();
    }
    if (t < 128) de[gi * 128 + f] = deacc;
}

// ---------- energy head tail: 128->64->1, per-atom ----------
__global__ __launch_bounds__(64) void head_kernel(
    const _Float16* __restrict__ h1, const float* __restrict__ W2,
    const float* __restrict__ b2, const float* __restrict__ W3,
    const float* __restrict__ b3, const float* __restrict__ AM,
    float* __restrict__ Ei)
{
    __shared__ float red[64];
    const int gi = blockIdx.x, n = threadIdx.x;
    float s = b2[n];
    for (int k = 0; k < 128; k++) s += (float)h1[gi * 128 + k] * W2[k * 64 + n];
    red[n] = silu_f(s) * W3[n];
    __syncthreads();
    for (int st = 32; st > 0; st >>= 1) {
        if (n < st) red[n] += red[n + st];
        __syncthreads();
    }
    if (n == 0) Ei[gi] = (red[0] + b3[0]) * AM[gi];
}

__global__ __launch_bounds__(256) void esum_kernel(const float* __restrict__ Ei,
                                                   float* __restrict__ Eout, int na) {
    __shared__ float red[256];
    const int b = blockIdx.x, t = threadIdx.x;
    float s = 0.f;
    for (int i = t; i < na; i += 256) s += Ei[b * na + i];
    red[t] = s;
    __syncthreads();
    for (int st = 128; st > 0; st >>= 1) {
        if (t < st) red[t] += red[t + st];
        __syncthreads();
    }
    if (t == 0) Eout[b] = red[0];
}

// ---------- host orchestration ----------
extern "C" void kernel_launch(void* const* d_in, const int* in_sizes, int n_in,
                              void* d_out, int out_size, void* d_ws, size_t ws_size,
                              hipStream_t stream)
{
    (void)in_sizes; (void)n_in; (void)out_size; (void)ws_size;
    const int NA = 2048, na = 512, NNB = 64, E = NA * NNB;

    const int*   Z   = (const int*)d_in[0];
    const float* D   = (const float*)d_in[1];
    const float* V   = (const float*)d_in[2];
    const int*   N   = (const int*)d_in[3];
    const float* NM  = (const float*)d_in[4];
    const float* AM  = (const float*)d_in[5];
    const float* emb = (const float*)d_in[6];

    char* ws = (char*)d_ws; size_t off = 0;
    auto carve = [&](size_t bytes) -> char* {
        off = (off + 255) & ~(size_t)255;
        char* p = ws + off; off += bytes; return p;
    };
    _Float16* wt    = (_Float16*)carve(31UL * 16384 * 2);
    float*    a     = (float*)carve((size_t)NA * 128 * 4);
    float*    a_m   = (float*)carve((size_t)NA * 128 * 4);
    _Float16* hn    = (_Float16*)carve((size_t)NA * 128 * 2);
    float*    phir  = (float*)carve((size_t)NA * 128 * 4);
    float*    de    = (float*)carve((size_t)NA * 128 * 4);
    float*    fsc   = (float*)carve((size_t)E * 4);
    float*    Fi    = (float*)carve((size_t)NA * 384 * 4);
    float*    fdyn  = (float*)carve((size_t)NA * 384 * 4);
    float*    r0    = (float*)carve((size_t)NA * 384 * 4);
    float*    r1    = (float*)carve((size_t)NA * 384 * 4);
    float*    EiB   = (float*)carve((size_t)NA * 4);
    _Float16* msij  = (_Float16*)carve((size_t)E * 128 * 2);
    _Float16* hedge = (_Float16*)carve((size_t)E * 128 * 2);
    _Float16* fsb   = (_Float16*)carve((size_t)E * 128 * 2);

    // convert+transpose the ten 128x128 mats per layer + ae.W1
    const int wsel[10] = {2, 4, 7, 9, 11, 13, 15, 16, 17, 19}; // aW1,aW2,fsW1,fsW2,rW1,rW2,reW1,reW2,eW1,eW2
    for (int L = 0; L < 3; L++)
        for (int j = 0; j < 10; j++)
            cvtT<<<64, 256, 0, stream>>>((const float*)d_in[7 + L * 21 + wsel[j]],
                                         wt + (size_t)(L * 10 + j) * 16384);
    cvtT<<<64, 256, 0, stream>>>((const float*)d_in[70], wt + 30UL * 16384);

    float* Eout = (float*)d_out;
    float* fdir = (float*)d_out + 4;
    zero_f32<<<(4 + NA * 3 + 255) / 256, 256, 0, stream>>>((float*)d_out, 4 + NA * 3);
    zero_f32<<<(NA * 384 + 255) / 256, 256, 0, stream>>>(fdyn, NA * 384);
    zero_f32<<<(NA * 384 + 255) / 256, 256, 0, stream>>>(r0, NA * 384);
    embed_kernel<<<NA * 128 / 256, 256, 0, stream>>>(Z, emb, a);

    float* rold = r0; float* rnew = r1;
    for (int L = 0; L < 3; L++) {
        void* const* P = d_in + 7 + L * 21;
        const float* rbfW = (const float*)P[0];  const float* rbfb = (const float*)P[1];
        const float* ab1  = (const float*)P[3];  const float* ab2  = (const float*)P[5];
        const float* fW   = (const float*)P[6];
        const float* fsb1 = (const float*)P[8];  const float* fsb2 = (const float*)P[10];
        const float* rb1  = (const float*)P[12]; const float* rb2  = (const float*)P[14];
        const float* eb1  = (const float*)P[18]; const float* eb2  = (const float*)P[20];
        _Float16* wL = wt + (size_t)L * 10 * 16384;

        // a_m = silu(a@aW1+b)@aW2+b
        gemm128<0, 1><<<NA / 16, 256, 0, stream>>>(a, wL + 0 * 16384, ab1, hn, nullptr, nullptr);
        gemm128<1, 0><<<NA / 16, 256, 0, stream>>>(hn, wL + 1 * 16384, ab2, a_m, nullptr, nullptr);
        // msij + fW dot + f_dir
        msij_kernel<<<NA, 256, 0, stream>>>(D, V, N, NM, a_m, rbfW, rbfb, fW, msij, fsc, fdir, na);
        // fs MLP (edge-level, WMMA heavy)
        gemm128<1, 1><<<E / 16, 256, 0, stream>>>(msij, wL + 2 * 16384, fsb1, hedge, nullptr, nullptr);
        gemm128<1, 3><<<E / 16, 256, 0, stream>>>(hedge, wL + 3 * 16384, fsb2, fsb, nullptr, nullptr);
        fi_kernel<<<NA, 256, 0, stream>>>(fsb, fsc, V, NM, Fi, fdyn);
        // phir MLP
        gemm128<0, 1><<<NA / 16, 256, 0, stream>>>(a, wL + 4 * 16384, rb1, hn, nullptr, nullptr);
        gemm128<1, 0><<<NA / 16, 256, 0, stream>>>(hn, wL + 5 * 16384, rb2, phir, nullptr, nullptr);
        // re MLP (no bias)
        gemm128<1, 1><<<E / 16, 256, 0, stream>>>(msij, wL + 6 * 16384, nullptr, hedge, nullptr, nullptr);
        gemm128<1, 3><<<E / 16, 256, 0, stream>>>(hedge, wL + 7 * 16384, nullptr, fsb, nullptr, nullptr);
        // r_dyn update + de
        drext_kernel<<<NA, 256, 0, stream>>>(fsb, N, NM, rold, phir, Fi, fdyn, rnew, de, na);
        // a += (silu(a@eW1+b)@eW2+b) * de   (fused epilogue)
        gemm128<0, 1><<<NA / 16, 256, 0, stream>>>(a, wL + 8 * 16384, eb1, hn, nullptr, nullptr);
        gemm128<1, 2><<<NA / 16, 256, 0, stream>>>(hn, wL + 9 * 16384, eb2, a, a, de);
        float* tmp = rold; rold = rnew; rnew = tmp;
    }

    // energy head: h1 = silu(a@aeW1+b1) via WMMA, then 128->64->1 + deterministic sum
    gemm128<0, 1><<<NA / 16, 256, 0, stream>>>(a, wt + 30UL * 16384,
                                               (const float*)d_in[71], hn, nullptr, nullptr);
    head_kernel<<<NA, 64, 0, stream>>>(hn, (const float*)d_in[72], (const float*)d_in[73],
                                       (const float*)d_in[74], (const float*)d_in[75], AM, EiB);
    esum_kernel<<<4, 256, 0, stream>>>(EiB, Eout, na);
}
